// DistributedXTW_twoway_66288525247026
// MI455X (gfx1250) — matrix operand order repaired and checked by reading, not compiled, not run on hardware
//
#include <hip/hip_runtime.h>
#include <cstdint>
#include <cstddef>

// ---------------------------------------------------------------------------
// y = x @ W + b   (M=8192, K=2048, N=8192, fp32)
// bf16 hi/lo split GEMM (3x v_wmma_f32_16x16x32_bf16 per tile-k step),
// TDM (tensor_load_to_lds) double-buffered LDS pipeline, wave32.
// ---------------------------------------------------------------------------

#define TOKENS  8192
#define IN_DIM  2048
#define OUT_DIM 8192

typedef __attribute__((ext_vector_type(16))) __bf16       v16bf;
typedef __attribute__((ext_vector_type(8)))  float        v8f;
typedef __attribute__((ext_vector_type(4)))  __bf16       bf16x4;
typedef __attribute__((ext_vector_type(4)))  unsigned int u32x4;
typedef __attribute__((ext_vector_type(8)))  int          i32x8;
typedef __attribute__((ext_vector_type(4)))  int          i32x4;

// ------------------------- conversion kernels ------------------------------

// Split x (M x K, row-major fp32) into bf16 hi/lo, same layout.
__global__ __launch_bounds__(256)
void k_split_x(const float* __restrict__ x,
               __bf16* __restrict__ hi, __bf16* __restrict__ lo) {
  const size_t i = (size_t)blockIdx.x * blockDim.x + threadIdx.x;
  const float4 v = ((const float4*)x)[i];
  bf16x4 h, l;
  h.x = (__bf16)v.x; l.x = (__bf16)(v.x - (float)h.x);
  h.y = (__bf16)v.y; l.y = (__bf16)(v.y - (float)h.y);
  h.z = (__bf16)v.z; l.z = (__bf16)(v.z - (float)h.z);
  h.w = (__bf16)v.w; l.w = (__bf16)(v.w - (float)h.w);
  ((bf16x4*)hi)[i] = h;
  ((bf16x4*)lo)[i] = l;
}

// Split + transpose W: (K x N fp32, row-major) -> Wt (N x K bf16 hi/lo).
__global__ __launch_bounds__(256)
void k_split_wT(const float* __restrict__ W,
                __bf16* __restrict__ wh, __bf16* __restrict__ wl) {
  __shared__ float tile[32][33];
  const int n0 = blockIdx.x * 32, k0 = blockIdx.y * 32;
  const int tx = threadIdx.x, ty = threadIdx.y;   // block (32,8)
#pragma unroll
  for (int i = 0; i < 32; i += 8)
    tile[ty + i][tx] = W[(size_t)(k0 + ty + i) * OUT_DIM + n0 + tx];
  __syncthreads();
#pragma unroll
  for (int i = 0; i < 32; i += 8) {
    const int n = n0 + ty + i;
    const int k = k0 + tx;
    const float v = tile[tx][ty + i];             // = W[k][n]
    const __bf16 h = (__bf16)v;
    const __bf16 l = (__bf16)(v - (float)h);
    const size_t o = (size_t)n * IN_DIM + k;
    wh[o] = h;
    wl[o] = l;
  }
}

// ------------------------------ GEMM kernel --------------------------------

#define KCH    32              // K elements per pipeline stage
#define ROWB   80              // padded LDS row stride: 64B data + 16B TDM pad
#define TILEB  (128 * ROWB)    // 10240 B per 128x32 bf16 tile
#define STAGEB (4 * TILEB)     // A_hi, A_lo, B_hi, B_lo

__device__ __forceinline__ unsigned lds_off_of(const void* p) {
  // Flat LDS address: aperture in [63:32], LDS byte offset in [31:0].
  return (unsigned)(unsigned long long)p;
}

// One TDM 2-D tile load: 128 rows x 32 bf16 from an (8192 x 2048) bf16
// tensor into LDS, with 16B padding inserted after every 64B row.
__device__ __forceinline__ void tdm_load_tile(unsigned lds_off,
                                              const __bf16* gsrc) {
  const unsigned long long ga = (unsigned long long)(const void*)gsrc;
  // D# group 0: count=1 | lds_addr | global_addr | type=2
  u32x4 g0 = { 1u,
               lds_off,
               (unsigned)(ga & 0xffffffffu),
               (unsigned)((ga >> 32) & 0x1ffffffu) | (2u << 30) };
  // D# group 1: data_size=2B, pad_enable, pad_interval=16DW(64B),
  // pad_amount=4DW(16B); tensor_dim0=K, tensor_dim1=8192,
  // tile_dim0=32, tile_dim1=128, dim0_stride=K.
  i32x8 g1 = { (int)((1u << 16) | (1u << 20) | (3u << 22) | (3u << 25)),
               (int)(((unsigned)IN_DIM & 0xffffu) << 16),
               (int)((((unsigned)IN_DIM >> 16) & 0xffffu) |
                     (((unsigned)TOKENS & 0xffffu) << 16)),
               (int)((((unsigned)TOKENS >> 16) & 0xffffu) | ((unsigned)KCH << 16)),
               (int)128,
               (int)IN_DIM,
               0, 0 };
  i32x4 z4 = { 0, 0, 0, 0 };
#if defined(__clang_major__) && (__clang_major__ >= 23)
  i32x8 z8 = { 0, 0, 0, 0, 0, 0, 0, 0 };
  __builtin_amdgcn_tensor_load_to_lds(g0, g1, z4, z4, z8, 0);
#else
  __builtin_amdgcn_tensor_load_to_lds(g0, g1, z4, z4, 0);
#endif
}

__device__ __forceinline__ v16bf ld_frag(const char* p, int off2) {
  union { uint4 q[2]; v16bf v; } u;
  u.q[0] = *(const uint4*)(p);
  u.q[1] = *(const uint4*)(p + off2);
  return u.v;
}

__device__ __forceinline__ v8f wmma_bf16(v16bf a, v16bf b, v8f c) {
  return __builtin_amdgcn_wmma_f32_16x16x32_bf16(
      /*neg_a=*/false, a, /*neg_b=*/false, b,
      /*c_mod=*/(short)0, c, /*reuse_a=*/false, /*reuse_b=*/false);
}

__global__ __launch_bounds__(256)
void k_gemm_bf16x3(const __bf16* __restrict__ xh, const __bf16* __restrict__ xl,
                   const __bf16* __restrict__ wh, const __bf16* __restrict__ wl,
                   const float* __restrict__ bias, float* __restrict__ out) {
  __shared__ char smem[2 * STAGEB];          // 80 KB, double buffered

  const int tid   = threadIdx.x;
  const int lane  = tid & 31;
  const int wave  = tid >> 5;                // 8 waves
  const int lrow  = lane & 15;
  const int lhalf = lane >> 4;
  const int bm = blockIdx.y * 128;           // block M origin
  const int bn = blockIdx.x * 128;           // block N origin
  const int wm = (wave & 1) * 64;            // wave M offset within block
  const int wn = (wave >> 1) * 32;           // wave N offset within block

  const v8f vzero = { 0.f, 0.f, 0.f, 0.f, 0.f, 0.f, 0.f, 0.f };
  v8f acc[4][2];
#pragma unroll
  for (int i = 0; i < 4; ++i)
#pragma unroll
    for (int j = 0; j < 2; ++j) acc[i][j] = vzero;

  const unsigned lds0 = lds_off_of(smem);
  const __bf16* gx_h = xh + (size_t)bm * IN_DIM;
  const __bf16* gx_l = xl + (size_t)bm * IN_DIM;
  const __bf16* gw_h = wh + (size_t)bn * IN_DIM;   // Wt is N x K
  const __bf16* gw_l = wl + (size_t)bn * IN_DIM;

  // Prologue: wave 0 DMAs stage 0 (TDM ignores EXEC; one issue per call).
  if (tid < 32) {
    tdm_load_tile(lds0 + 0 * TILEB, gx_h);
    tdm_load_tile(lds0 + 1 * TILEB, gx_l);
    tdm_load_tile(lds0 + 2 * TILEB, gw_h);
    tdm_load_tile(lds0 + 3 * TILEB, gw_l);
  }

  const int KB = IN_DIM / KCH;               // 64 stages
  for (int kb = 0; kb < KB; ++kb) {
    const int s = kb & 1;
    if (tid < 32) {
      if (kb + 1 < KB) {
        const unsigned so = lds0 + (s ^ 1) * STAGEB;
        const int ko = (kb + 1) * KCH;
        tdm_load_tile(so + 0 * TILEB, gx_h + ko);
        tdm_load_tile(so + 1 * TILEB, gx_l + ko);
        tdm_load_tile(so + 2 * TILEB, gw_h + ko);
        tdm_load_tile(so + 3 * TILEB, gw_l + ko);
        __builtin_amdgcn_s_wait_tensorcnt(4);  // stage kb (older 4) done
      } else {
        __builtin_amdgcn_s_wait_tensorcnt(0);
      }
    }
    __syncthreads();

    const char* Ah = smem + s * STAGEB;
    const char* Al = Ah + TILEB;
    const char* Bh = Ah + 2 * TILEB;
    const char* Bl = Ah + 3 * TILEB;

    // Fragment loads per ISA VGPR layouts (two ds_load_b128 each).
    v16bf afh[4], afl[4], bfh[2], bfl[2];
#pragma unroll
    for (int i = 0; i < 4; ++i) {
      const int off = (wm + i * 16 + lrow) * ROWB + lhalf * 16;
      afh[i] = ld_frag(Ah + off, 32);
      afl[i] = ld_frag(Al + off, 32);
    }
#pragma unroll
    for (int j = 0; j < 2; ++j) {
      const int off = (wn + j * 16 + lrow) * ROWB + lhalf * 16;
      bfh[j] = ld_frag(Bh + off, 32);
      bfl[j] = ld_frag(Bl + off, 32);
    }

    // 3-term bf16 split: hi*hi + hi*lo + lo*hi, fp32 accumulate.
#pragma unroll
    for (int i = 0; i < 4; ++i)
#pragma unroll
      for (int j = 0; j < 2; ++j) {
        acc[i][j] = wmma_bf16(afh[i], bfh[j], acc[i][j]);
        acc[i][j] = wmma_bf16(afh[i], bfl[j], acc[i][j]);
        acc[i][j] = wmma_bf16(afl[i], bfh[j], acc[i][j]);
      }
    __syncthreads();  // buffer s free before it is re-issued next iteration
  }

  // Epilogue: bias add + store (C/D layout: N=lane&15, M=vgpr + 8*(lane>>4)).
#pragma unroll
  for (int j = 0; j < 2; ++j) {
    const int n = bn + wn + j * 16 + lrow;
    const float bv = bias[n];
#pragma unroll
    for (int i = 0; i < 4; ++i) {
      const int m0 = bm + wm + i * 16 + lhalf * 8;
#pragma unroll
      for (int v = 0; v < 8; ++v)
        out[(size_t)(m0 + v) * OUT_DIM + n] = acc[i][j][v] + bv;
    }
  }
}

// ------------------------------ launcher -----------------------------------

extern "C" void kernel_launch(void* const* d_in, const int* in_sizes, int n_in,
                              void* d_out, int out_size, void* d_ws, size_t ws_size,
                              hipStream_t stream) {
  (void)in_sizes; (void)n_in; (void)out_size; (void)ws_size;

  const float* x = (const float*)d_in[0];   // (8192, 2048)
  const float* W = (const float*)d_in[1];   // (2048, 8192)
  const float* b = (const float*)d_in[2];   // (8192,)
  float* out = (float*)d_out;               // (8192, 8192)

  const size_t XEL = (size_t)TOKENS * IN_DIM;     // 16,777,216 elements
  const size_t WEL = (size_t)IN_DIM * OUT_DIM;    // 16,777,216 elements

  // Workspace: x_hi | x_lo | Wt_hi | Wt_lo  (32 MB each, 128 MB total)
  __bf16* xh = (__bf16*)d_ws;
  __bf16* xl = xh + XEL;
  __bf16* wh = xl + XEL;
  __bf16* wl = wh + WEL;

  // 1) split x -> bf16 hi/lo (row-major M x K)
  k_split_x<<<(unsigned)(XEL / 4 / 256), 256, 0, stream>>>(x, xh, xl);

  // 2) transpose + split W: (K x N) -> Wt (N x K) bf16 hi/lo
  k_split_wT<<<dim3(OUT_DIM / 32, IN_DIM / 32), dim3(32, 8), 0, stream>>>(W, wh, wl);

  // 3) TDM-pipelined bf16x3 WMMA GEMM + bias
  k_gemm_bf16x3<<<dim3(OUT_DIM / 128, TOKENS / 128), 256, 0, stream>>>(
      xh, xl, wh, wl, b, out);
}
